// SparseMoE_63161789054987
// MI455X (gfx1250) — compile-verified, matrix operand order checked by reading, not run on hardware
//
#include <hip/hip_runtime.h>
#include <hip/hip_bf16.h>
#include <math.h>

// ---------------- problem constants (match reference) ----------------
#define D_MODEL   1024
#define FFN_DIM   4096
#define N_EXPERTS 8
#define T_TOKENS  8192      // B*L
#define M_TILE    32        // tokens per FFN workgroup (2 WMMA M tiles)
#define KC        128       // ffn-dimension chunk per fused step
#define XP        1032      // x_lds pitch (bf16): 16B-aligned rows, bank-spread
#define HP2       136       // h_lds pitch (bf16), chunk is 128 wide
#define PW        136       // staged weight block pitch: [128 n][128 k] transposed

typedef __bf16 bf16_t;
typedef bf16_t v16bf  __attribute__((ext_vector_type(16)));
typedef bf16_t bf16x2 __attribute__((ext_vector_type(2)));
typedef bf16_t bf16x4 __attribute__((ext_vector_type(4)));
typedef float  v8f    __attribute__((ext_vector_type(8)));

__device__ __forceinline__ bf16_t f2bf(float f) { return (bf16_t)f; }

// 16xK32 bf16 fragment from LDS, row-major [r][k] with pitch (elems).
// CDNA5 16-bit layout: lanes 0-15 -> r=lane, K {0..7,16..23}; lanes 16-31 ->
// r=lane-16, K {8..15,24..31}. 8-element K groups contiguous -> 2x ds_load_b128.
__device__ __forceinline__ v16bf ldsFrag(const bf16_t* base, int pitch, int k0) {
  const int lane = threadIdx.x & 31;
  const bf16_t* p = base + (lane & 15) * pitch + k0 + ((lane & 16) ? 8 : 0);
  v16bf a;
#pragma unroll
  for (int i = 0; i < 8; ++i) a[i] = p[i];
#pragma unroll
  for (int i = 0; i < 8; ++i) a[8 + i] = p[16 + i];
  return a;
}

__device__ __forceinline__ v8f wmma_bf16(v16bf a, v16bf b, v8f c) {
  return __builtin_amdgcn_wmma_f32_16x16x32_bf16(false, a, false, b,
                                                 (short)0, c, false, false);
}

// ---- weight block staging: [128 k][128 n] fp32 -> LDS [n][k] bf16, K-paired.
// Split into load / store halves so global-load latency hides under WMMAs.
__device__ __forceinline__ void stage_load(const float* __restrict__ W, int ld,
                                           int krow0, int n0, float4* r, int tid) {
#pragma unroll
  for (int i = 0; i < 8; ++i) {
    const int q  = tid + i * 256;          // 0..2047
    const int kp = q >> 5;                 // k pair 0..63
    const int c4 = (q & 31) * 4;           // n_local 0..124
    const float* p = W + (size_t)(krow0 + kp * 2) * ld + n0 + c4;
    r[2 * i]     = *(const float4*)p;
    r[2 * i + 1] = *(const float4*)(p + ld);
  }
}
__device__ __forceinline__ void stage_store(bf16_t* buf, const float4* r, int tid) {
#pragma unroll
  for (int i = 0; i < 8; ++i) {
    const int q  = tid + i * 256;
    const int kp = q >> 5;
    const int c4 = (q & 31) * 4;
    const float4 r0 = r[2 * i], r1 = r[2 * i + 1];
    bf16_t* d = buf + c4 * PW + kp * 2;
    *(bf16x2*)(d)          = (bf16x2){f2bf(r0.x), f2bf(r1.x)};
    *(bf16x2*)(d + PW)     = (bf16x2){f2bf(r0.y), f2bf(r1.y)};
    *(bf16x2*)(d + 2 * PW) = (bf16x2){f2bf(r0.z), f2bf(r1.z)};
    *(bf16x2*)(d + 3 * PW) = (bf16x2){f2bf(r0.w), f2bf(r1.w)};
  }
}

// ---------------- kernel 1: router (top-2 softmax + expert gather lists) ----
__global__ __launch_bounds__(256) void moe_router(
    const float* __restrict__ x, const float* __restrict__ gate_w,
    const float* __restrict__ gate_b, int* __restrict__ cnt,
    int* __restrict__ toks, float* __restrict__ gts) {
  __shared__ float gw_s[D_MODEL * N_EXPERTS];
  __shared__ float gb_s[N_EXPERTS];
  const int tid = threadIdx.x;
  for (int i = tid; i < D_MODEL * N_EXPERTS; i += 256) gw_s[i] = gate_w[i];
  if (tid < N_EXPERTS) gb_s[tid] = gate_b[tid];
  __syncthreads();

  const int t = blockIdx.x * 256 + tid;
  const float* xr = x + (size_t)t * D_MODEL;
  float lg[N_EXPERTS];
#pragma unroll
  for (int e = 0; e < N_EXPERTS; ++e) lg[e] = gb_s[e];
  for (int k = 0; k < D_MODEL; k += 4) {
    const float4 xv = *(const float4*)(xr + k);
    const float xs[4] = {xv.x, xv.y, xv.z, xv.w};
#pragma unroll
    for (int j = 0; j < 4; ++j)
#pragma unroll
      for (int e = 0; e < N_EXPERTS; ++e)
        lg[e] = fmaf(xs[j], gw_s[(k + j) * N_EXPERTS + e], lg[e]);
  }
  int e0 = 0; float v0 = lg[0];
#pragma unroll
  for (int e = 1; e < N_EXPERTS; ++e) if (lg[e] > v0) { v0 = lg[e]; e0 = e; }
  int e1 = (e0 == 0) ? 1 : 0; float v1 = lg[e1];
#pragma unroll
  for (int e = 0; e < N_EXPERTS; ++e)
    if (e != e0 && lg[e] > v1) { v1 = lg[e]; e1 = e; }
  const float ex = __expf(v1 - v0);
  const float inv = 1.0f / (1.0f + ex);
  const float g0 = inv, g1 = ex * inv;

  int p0 = atomicAdd(&cnt[e0], 1);
  toks[e0 * T_TOKENS + p0] = t; gts[e0 * T_TOKENS + p0] = g0;
  int p1 = atomicAdd(&cnt[e1], 1);
  toks[e1 * T_TOKENS + p1] = t; gts[e1 * T_TOKENS + p1] = g1;
}

// ---------------- kernel 2: fused gathered expert FFN ----------------------
// grid = (T_TOKENS/M_TILE, N_EXPERTS); 256 threads = 8 waves (wave32).
// Chunked over FFN in KC=128 columns. Wave w owns n-column w*16 of each staged
// 128-wide block and computes BOTH M sub-tiles against each B fragment.
// Weight staging is double-buffered: next block's global loads are issued
// before the current block's WMMAs; convert+store after; one barrier per block.
__global__ __launch_bounds__(256) void moe_ffn(
    const float* __restrict__ x,  const float* __restrict__ w1,
    const float* __restrict__ b1, const float* __restrict__ w2,
    const float* __restrict__ b2, const int* __restrict__ cnt,
    const int* __restrict__ toks, const float* __restrict__ gts,
    float* __restrict__ out) {
  __shared__ bf16_t x_s[M_TILE * XP];          // 66.0 KB
  __shared__ bf16_t wbuf_s[2][128 * PW];       // 2 x 34.8 KB ping-pong
  __shared__ bf16_t h_s[M_TILE * HP2];         //  8.7 KB
  __shared__ int    tok_s[M_TILE];
  __shared__ float  gate_s[M_TILE];

  const int e = blockIdx.y;
  const int count = cnt[e];
  const int base = blockIdx.x * M_TILE;
  if (base >= count) return;

  const int tid = threadIdx.x;
  if (tid < M_TILE) {
    const int idx = base + tid;
    const bool ok = idx < count;
    tok_s[tid]  = ok ? toks[e * T_TOKENS + idx] : 0;
    gate_s[tid] = ok ? gts[e * T_TOKENS + idx] : 0.0f;   // gate 0 => no effect
  }
  __syncthreads();

  // stage gathered x rows: float4 global loads -> packed bf16x4 LDS stores
#pragma unroll
  for (int i = 0; i < (M_TILE * D_MODEL) / (4 * 256); ++i) {
    const int q = tid + i * 256;
    const int r = q >> 8, c = (q & 255) * 4;
    const float4 xv = *(const float4*)(x + (size_t)tok_s[r] * D_MODEL + c);
    bf16x4 p = {f2bf(xv.x), f2bf(xv.y), f2bf(xv.z), f2bf(xv.w)};
    *(bf16x4*)(x_s + r * XP + c) = p;
  }
  __syncthreads();

  const float* w1e = w1 + (size_t)e * D_MODEL * FFN_DIM;
  const float* w2e = w2 + (size_t)e * FFN_DIM * D_MODEL;
  const float* b1e = b1 + (size_t)e * FFN_DIM;
  const float* b2e = b2 + (size_t)e * D_MODEL;

  const int wv   = tid >> 5;
  const int lane = tid & 31;
  const int lhi  = (lane & 16) ? 8 : 0;
  const int ln   = lane & 15;

  v8f yacc[16];
#pragma unroll
  for (int t = 0; t < 16; ++t)
    yacc[t] = (v8f){0.f, 0.f, 0.f, 0.f, 0.f, 0.f, 0.f, 0.f};

  float4 sreg[16];   // staging pipeline registers (16 x float4)

  for (int fc = 0; fc < FFN_DIM; fc += KC) {
    // ======= GEMM1: h[32 x 128] at ffn cols [fc, fc+128), K = D_MODEL =======
    v8f h0 = (v8f){0.f, 0.f, 0.f, 0.f, 0.f, 0.f, 0.f, 0.f};
    v8f h1 = h0;
    stage_load(w1e, FFN_DIM, 0, fc, sreg, tid);
    stage_store(wbuf_s[0], sreg, tid);
    __syncthreads();
    for (int kb = 0; kb < D_MODEL; kb += 128) {
      const int cur = (kb >> 7) & 1;
      const bool more = (kb + 128) < D_MODEL;
      if (more) stage_load(w1e, FFN_DIM, kb + 128, fc, sreg, tid);
      const bf16_t* buf = wbuf_s[cur] + wv * 16 * PW;
#pragma unroll
      for (int k4 = 0; k4 < 128; k4 += 32) {
        v16bf b  = ldsFrag(buf, PW, k4);
        v16bf a0 = ldsFrag(x_s, XP, kb + k4);
        v16bf a1 = ldsFrag(x_s + 16 * XP, XP, kb + k4);
        h0 = wmma_bf16(a0, b, h0);
        h1 = wmma_bf16(a1, b, h1);
      }
      if (more) stage_store(wbuf_s[cur ^ 1], sreg, tid);
      __syncthreads();
    }
    // exact GELU + bf16 h store (rows 0..31, chunk-local col wv*16+ln)
    const float bias = b1e[fc + wv * 16 + ln];
#pragma unroll
    for (int r = 0; r < 8; ++r) {
      float v = h0[r] + bias;
      v = 0.5f * v * (1.0f + erff(v * 0.70710678118654752f));
      h_s[(r + lhi) * HP2 + wv * 16 + ln] = f2bf(v);
      float u = h1[r] + bias;
      u = 0.5f * u * (1.0f + erff(u * 0.70710678118654752f));
      h_s[(16 + r + lhi) * HP2 + wv * 16 + ln] = f2bf(u);
    }
    __syncthreads();

    // ======= GEMM2 partial: yacc += h_chunk @ w2[fc:fc+128, :] =======
    stage_load(w2e, D_MODEL, fc, 0, sreg, tid);
    stage_store(wbuf_s[0], sreg, tid);
    __syncthreads();
    for (int nb = 0; nb < 8; ++nb) {
      const int cur = nb & 1;
      if (nb < 7) stage_load(w2e, D_MODEL, fc, (nb + 1) * 128, sreg, tid);
      const bf16_t* buf = wbuf_s[cur] + wv * 16 * PW;
      v8f acc0 = yacc[nb * 2], acc1 = yacc[nb * 2 + 1];
#pragma unroll
      for (int kk = 0; kk < 128; kk += 32) {
        v16bf b  = ldsFrag(buf, PW, kk);
        v16bf a0 = ldsFrag(h_s, HP2, kk);
        v16bf a1 = ldsFrag(h_s + 16 * HP2, HP2, kk);
        acc0 = wmma_bf16(a0, b, acc0);
        acc1 = wmma_bf16(a1, b, acc1);
      }
      yacc[nb * 2] = acc0; yacc[nb * 2 + 1] = acc1;
      if (nb < 7) stage_store(wbuf_s[cur ^ 1], sreg, tid);
      __syncthreads();
    }
  }

  // ---- epilogue: out[token] += gate * (y + b2), top-2 combine via atomics
#pragma unroll
  for (int nb = 0; nb < 8; ++nb)
#pragma unroll
    for (int m = 0; m < 2; ++m) {
      const int n = nb * 128 + wv * 16 + ln;
      const float bv = b2e[n];
#pragma unroll
      for (int r = 0; r < 8; ++r) {
        const int row = m * 16 + r + lhi;
        const float val = gate_s[row] * (yacc[nb * 2 + m][r] + bv);
        atomicAdd(&out[(size_t)tok_s[row] * D_MODEL + n], val);
      }
    }
}

// ---------------- launch ---------------------------------------------------
extern "C" void kernel_launch(void* const* d_in, const int* in_sizes, int n_in,
                              void* d_out, int out_size, void* d_ws, size_t ws_size,
                              hipStream_t stream) {
  const float* x      = (const float*)d_in[0];
  const float* gate_w = (const float*)d_in[1];
  const float* gate_b = (const float*)d_in[2];
  const float* w1     = (const float*)d_in[3];
  const float* b1     = (const float*)d_in[4];
  const float* w2     = (const float*)d_in[5];
  const float* b2     = (const float*)d_in[6];
  float* out = (float*)d_out;

  int*   cnt  = (int*)d_ws;
  int*   toks = (int*)((char*)d_ws + 1024);
  float* gts  = (float*)((char*)d_ws + 1024 + (size_t)N_EXPERTS * T_TOKENS * 4);

  hipMemsetAsync(d_out, 0, (size_t)out_size * sizeof(float), stream);
  hipMemsetAsync(cnt, 0, N_EXPERTS * sizeof(int), stream);

  moe_router<<<T_TOKENS / 256, 256, 0, stream>>>(x, gate_w, gate_b, cnt, toks, gts);

  dim3 grid(T_TOKENS / M_TILE, N_EXPERTS);
  moe_ffn<<<grid, 256, 0, stream>>>(x, w1, b1, w2, b2, cnt, toks, gts, out);
}
